// SparseGRU_25168508354650
// MI455X (gfx1250) — matrix operand order repaired and checked by reading, not compiled, not run on hardware
//
#include <hip/hip_runtime.h>
#include <hip/hip_bf16.h>

typedef __bf16 bf16;
typedef __attribute__((ext_vector_type(16))) __bf16 v16bf;
typedef __attribute__((ext_vector_type(8)))  __bf16 v8bf;
typedef __attribute__((ext_vector_type(8)))  float  v8f;

#define B_SZ 64
#define T_SZ 512
#define I_SZ 512
#define H_SZ 1024
#define G3   (3 * H_SZ)      // 3072
#define NGT  (G3 / 16)       // 192 g-tiles (16 output rows each)
#define NB_IH (I_SZ / 32)    // 16  K32-blocks in input dim
#define NB_HH (H_SZ / 32)    // 32  K32-blocks in hidden dim
#define NBLOCKS 256          // persistent scan grid (4 b-tiles x 64 h-tiles)

union ABfrag { v16bf v; v8bf h[2]; };

__device__ __forceinline__ bf16 f2bf(float f) {
    unsigned u = __builtin_bit_cast(unsigned, f);
    unsigned r = u + 0x7FFFu + ((u >> 16) & 1u);   // round-to-nearest-even
    unsigned short hs = (unsigned short)(r >> 16);
    return __builtin_bit_cast(bf16, hs);
}

// A fragment (16x32 bf16, M = batch rows): lane r=lane&15 owns row r;
// lanes<16 hold K = k0+[0..15], lanes>=16 hold K = k0+[16..31].
__device__ __forceinline__ v16bf load_afrag(const bf16* __restrict__ row,
                                            int k0, int hi) {
    ABfrag a;
    a.h[0] = *(const v8bf*)(row + k0 + 16 * hi);
    a.h[1] = *(const v8bf*)(row + k0 + 16 * hi + 8);
    return a.v;
}

// ---------------------------------------------------------------------------
// f32 -> bf16 cast (grid-stride)
// ---------------------------------------------------------------------------
__global__ void cast_f32_bf16(const float* __restrict__ src,
                              bf16* __restrict__ dst, int n) {
    int i = blockIdx.x * blockDim.x + threadIdx.x;
    int stride = gridDim.x * blockDim.x;
    for (; i < n; i += stride) dst[i] = f2bf(src[i]);
}

// ---------------------------------------------------------------------------
// zero-init h ping buffers + grid-barrier counters
// ---------------------------------------------------------------------------
__global__ void init_state(float* __restrict__ hf, bf16* __restrict__ hb,
                           unsigned* __restrict__ bar, int n) {
    int i = blockIdx.x * blockDim.x + threadIdx.x;
    if (i < n) { hf[i] = 0.0f; hb[i] = f2bf(0.0f); }
    if (i == 0) { bar[0] = 0u; bar[1] = 0u; }
}

// ---------------------------------------------------------------------------
// Build nonzero K32-block lists per 16-row g-tile (deterministic, no atomics).
// One wave per g-tile; lane kb scans the 16 x 32 block of w for any nonzero.
// ---------------------------------------------------------------------------
__global__ __launch_bounds__(32)
void build_nz(const float* __restrict__ w, int kdim, int nkb,
              unsigned* __restrict__ nzc, unsigned* __restrict__ nzl,
              int lstride) {
    const int gt   = blockIdx.x;
    const int lane = threadIdx.x;
    unsigned nz = 0;
    if (lane < nkb) {
        const float* base = w + (size_t)gt * 16 * kdim + lane * 32;
        for (int row = 0; row < 16 && !nz; ++row) {
            const float4* p = (const float4*)(base + (size_t)row * kdim);
#pragma unroll
            for (int c = 0; c < 8; ++c) {
                float4 v = p[c];
                if (v.x != 0.f || v.y != 0.f || v.z != 0.f || v.w != 0.f) nz = 1u;
            }
        }
    }
    unsigned flags = 0;                       // all lanes participate in shfl
    for (int i = 0; i < nkb; ++i)
        flags |= ((unsigned)__shfl((int)nz, i, 32)) << i;
    if (lane == 0) {
        unsigned cnt = 0;
        for (int i = 0; i < nkb; ++i)
            if ((flags >> i) & 1u) nzl[(size_t)gt * lstride + cnt++] = (unsigned)i;
        nzc[gt] = cnt;
    }
}

// ---------------------------------------------------------------------------
// gi_all[t][b][g] = sum_i x[b][t][i]*w_ih[g][i] + b_ih[g], skipping zero blocks
// One wave per 16-row x 64-col tile; row R=t*64+b so a tile has a single t.
// ---------------------------------------------------------------------------
__global__ __launch_bounds__(32)
void gemm_gi(const bf16* __restrict__ xb,    // [B,T,I] bf16
             const bf16* __restrict__ wih,   // [3H,I]  bf16
             const float* __restrict__ bih,  // [3H]
             const unsigned* __restrict__ nzc,
             const unsigned* __restrict__ nzl,
             float* __restrict__ gi)         // [T,B,3H] f32
{
    const int lane = threadIdx.x & 31;
    const int r    = lane & 15;
    const int hi   = lane >> 4;

    const int rt = blockIdx.x;          // 0..2047 row tiles
    const int cg = blockIdx.y;          // 0..47   64-col groups
    const int R0 = rt * 16;
    const int t  = R0 >> 6;
    const int b0 = R0 & 63;

    const bf16* arow = xb + ((size_t)(b0 + r) * T_SZ + t) * I_SZ;

    v8f acc[4] = {};
#pragma unroll
    for (int j = 0; j < 4; ++j) {
        const int gt = cg * 4 + j;
        const int g  = cg * 64 + j * 16 + r;         // this lane's B column
        const bf16* brow = wih + (size_t)g * I_SZ;
        const unsigned cnt = nzc[gt];
        const unsigned* lst = nzl + (size_t)gt * NB_IH;
        for (unsigned i = 0; i < cnt; ++i) {
            const int k0 = (int)lst[i] * 32;
            v16bf a = load_afrag(arow, k0, hi);
            v16bf b = *(const v16bf*)(brow + k0 + 16 * hi);
            acc[j] = __builtin_amdgcn_wmma_f32_16x16x32_bf16(
                false, a, false, b, (short)0, acc[j], false, false);
        }
    }

    const int n = lane & 15;
#pragma unroll
    for (int j = 0; j < 4; ++j) {
        const int g = cg * 64 + j * 16 + n;
        const float bias = bih[g];
#pragma unroll
        for (int v = 0; v < 8; ++v) {
            const int b = b0 + v + 8 * hi;           // C row = v + 8*(lane>=16)
            gi[((size_t)t * B_SZ + b) * G3 + g] = acc[j][v] + bias;
        }
    }
}

// ---------------------------------------------------------------------------
// Device-wide generation barrier (256 co-resident single-wave blocks).
// ---------------------------------------------------------------------------
__device__ __forceinline__ void grid_sync(unsigned* cnt, unsigned* gen) {
    __threadfence();                               // publish this block's writes
    if (threadIdx.x == 0) {
        unsigned g = __hip_atomic_load(gen, __ATOMIC_RELAXED, __HIP_MEMORY_SCOPE_AGENT);
        unsigned prev = __hip_atomic_fetch_add(cnt, 1u, __ATOMIC_ACQ_REL,
                                               __HIP_MEMORY_SCOPE_AGENT);
        if (prev == NBLOCKS - 1) {
            __hip_atomic_store(cnt, 0u, __ATOMIC_RELAXED, __HIP_MEMORY_SCOPE_AGENT);
            __hip_atomic_store(gen, g + 1u, __ATOMIC_RELEASE, __HIP_MEMORY_SCOPE_AGENT);
        } else {
            while (__hip_atomic_load(gen, __ATOMIC_ACQUIRE,
                                     __HIP_MEMORY_SCOPE_AGENT) == g)
                __builtin_amdgcn_s_sleep(2);
        }
    }
    __syncthreads();
}

// ---------------------------------------------------------------------------
// Persistent GRU scan: one launch, T internal steps, grid barrier between.
// Each wave owns a 16(batch) x 16(hidden) tile of all THREE gates.
// ---------------------------------------------------------------------------
__global__ __launch_bounds__(32)
void gru_scan(const bf16*  __restrict__ whh,   // [3H,H] bf16
              const float* __restrict__ bhh,   // [3H]
              const float* __restrict__ gi,    // [T,B,3H] f32
              float* __restrict__ hf0, float* __restrict__ hf1,
              bf16*  __restrict__ hb0, bf16*  __restrict__ hb1,
              const unsigned* __restrict__ nzc,
              const unsigned* __restrict__ nzl,
              float* __restrict__ out,         // [B,T,H] then [B,H]
              unsigned* __restrict__ bar)
{
    const int lane = threadIdx.x & 31;
    const int r    = lane & 15;
    const int hi   = lane >> 4;
    const int b0   = blockIdx.x * 16;       // 4 batch tiles
    const int h0   = blockIdx.y * 16;       // 64 hidden tiles
    const int ht   = h0 >> 4;

    const int hc = h0 + r;                  // this lane's hidden column (C: N=lane&15)
    const float bhr = bhh[hc];
    const float bhz = bhh[H_SZ + hc];
    const float bhn = bhh[2 * H_SZ + hc];
    float* hlast = out + (size_t)B_SZ * T_SZ * H_SZ;

    for (int t = 0; t < T_SZ; ++t) {
        const bf16*  hbs = (t & 1) ? hb1 : hb0;
        const float* hfs = (t & 1) ? hf1 : hf0;
        bf16*  hbd = (t & 1) ? hb0 : hb1;
        float* hfd = (t & 1) ? hf0 : hf1;
        const float* gi_t = gi + (size_t)t * B_SZ * G3;

        const bf16* arow = hbs + (size_t)(b0 + r) * H_SZ;
        __builtin_prefetch(arow, 0, 1);

        v8f acc[3];
#pragma unroll
        for (int gate = 0; gate < 3; ++gate) {
            acc[gate] = (v8f){};
            const int gt = gate * (H_SZ / 16) + ht;
            const int g  = gate * H_SZ + h0 + r;
            const bf16* brow = whh + (size_t)g * H_SZ;
            const unsigned cnt = nzc[gt];
            const unsigned* lst = nzl + (size_t)gt * NB_HH;
            for (unsigned i = 0; i < cnt; ++i) {
                const int k0 = (int)lst[i] * 32;
                v16bf a = load_afrag(arow, k0, hi);
                v16bf b = *(const v16bf*)(brow + k0 + 16 * hi);
                acc[gate] = __builtin_amdgcn_wmma_f32_16x16x32_bf16(
                    false, a, false, b, (short)0, acc[gate], false, false);
            }
        }

#pragma unroll
        for (int v = 0; v < 8; ++v) {
            const int b = b0 + v + 8 * hi;          // C row = v + 8*(lane>=16)
            const size_t gb = (size_t)b * G3;
            const float gi_r = gi_t[gb + hc];
            const float gi_z = gi_t[gb + H_SZ + hc];
            const float gi_n = gi_t[gb + 2 * H_SZ + hc];
            const float hprev = hfs[(size_t)b * H_SZ + hc];

            const float rg = 1.0f / (1.0f + __expf(-(gi_r + acc[0][v] + bhr)));
            const float zg = 1.0f / (1.0f + __expf(-(gi_z + acc[1][v] + bhz)));
            const float ng = tanhf(gi_n + rg * (acc[2][v] + bhn));
            const float hnew = (1.0f - zg) * ng + zg * hprev;

            hfd[(size_t)b * H_SZ + hc] = hnew;
            hbd[(size_t)b * H_SZ + hc] = f2bf(hnew);
            out[((size_t)b * T_SZ + t) * H_SZ + hc] = hnew;
            if (t == T_SZ - 1) hlast[(size_t)b * H_SZ + hc] = hnew;
        }

        grid_sync(&bar[0], &bar[1]);
    }
}

// ---------------------------------------------------------------------------
extern "C" void kernel_launch(void* const* d_in, const int* in_sizes, int n_in,
                              void* d_out, int out_size, void* d_ws, size_t ws_size,
                              hipStream_t stream) {
    const float* x    = (const float*)d_in[0];
    const float* w_ih = (const float*)d_in[1];
    const float* w_hh = (const float*)d_in[2];
    const float* b_ih = (const float*)d_in[3];
    const float* b_hh = (const float*)d_in[4];
    float* out = (float*)d_out;                 // [B,T,H] then [B,H]

    char* ws = (char*)d_ws;
    size_t off = 0;
    bf16* xb   = (bf16*)(ws + off); off += (size_t)B_SZ * T_SZ * I_SZ * 2;
    bf16* wihb = (bf16*)(ws + off); off += (size_t)G3 * I_SZ * 2;
    bf16* whhb = (bf16*)(ws + off); off += (size_t)G3 * H_SZ * 2;
    float* gi  = (float*)(ws + off); off += (size_t)T_SZ * B_SZ * G3 * 4;
    float* hf0 = (float*)(ws + off); off += (size_t)B_SZ * H_SZ * 4;
    float* hf1 = (float*)(ws + off); off += (size_t)B_SZ * H_SZ * 4;
    bf16*  hb0 = (bf16*)(ws + off); off += (size_t)B_SZ * H_SZ * 2;
    bf16*  hb1 = (bf16*)(ws + off); off += (size_t)B_SZ * H_SZ * 2;
    unsigned* nzc_ih = (unsigned*)(ws + off); off += NGT * 4;
    unsigned* nzl_ih = (unsigned*)(ws + off); off += (size_t)NGT * NB_IH * 4;
    unsigned* nzc_hh = (unsigned*)(ws + off); off += NGT * 4;
    unsigned* nzl_hh = (unsigned*)(ws + off); off += (size_t)NGT * NB_HH * 4;
    unsigned* bar    = (unsigned*)(ws + off); off += 2 * 4;

    cast_f32_bf16<<<2048, 256, 0, stream>>>(x,    xb,   B_SZ * T_SZ * I_SZ);
    cast_f32_bf16<<<512,  256, 0, stream>>>(w_ih, wihb, G3 * I_SZ);
    cast_f32_bf16<<<512,  256, 0, stream>>>(w_hh, whhb, G3 * H_SZ);

    build_nz<<<NGT, 32, 0, stream>>>(w_ih, I_SZ, NB_IH, nzc_ih, nzl_ih, NB_IH);
    build_nz<<<NGT, 32, 0, stream>>>(w_hh, H_SZ, NB_HH, nzc_hh, nzl_hh, NB_HH);

    init_state<<<(B_SZ * H_SZ) / 256, 256, 0, stream>>>(hf0, hb0, bar, B_SZ * H_SZ);

    // input-side GEMM over nonzero blocks only
    gemm_gi<<<dim3(2048, 48), 32, 0, stream>>>(xb, wihb, b_ih, nzc_ih, nzl_ih, gi);

    // persistent recurrent scan: single launch, internal grid barrier per step
    gru_scan<<<dim3(4, 64), 32, 0, stream>>>(whhb, b_hh, gi, hf0, hf1, hb0, hb1,
                                             nzc_hh, nzl_hh, out, bar);
}
